// BioScaleGNN_33569464386145
// MI455X (gfx1250) — matrix operand order repaired and checked by reading, not compiled, not run on hardware
//
#include <hip/hip_runtime.h>
#include <hip/hip_bf16.h>

typedef __attribute__((ext_vector_type(16))) _Float16 v16h;
typedef __attribute__((ext_vector_type(8)))  float    v8f;

#define N_NODES 8192
#define IN_DIM  128
#define HID     256
#define OUT_DIM 128
#define QKV_DIM 768
#define N_HEADS 4
#define HEAD_DIM 64
#define NEDGE   524288

#define WMMA_F16(a, b, c) \
  __builtin_amdgcn_wmma_f32_16x16x32_f16(false, (a), false, (b), (short)0, (c), false, false)

// ---- fragment loaders (f32 memory -> f16 WMMA fragments) ---------------------
// B-fragment: 16 contiguous K values per lane (lanes 0-15: K 0-15, lanes 16-31:
// K 16-31; caller adds hi*16 to the pointer). Works for global or LDS pointers.
__device__ __forceinline__ v16h ld16f(const float* __restrict__ p) {
  const float4 a = *(const float4*)(p + 0);
  const float4 b = *(const float4*)(p + 4);
  const float4 c = *(const float4*)(p + 8);
  const float4 d = *(const float4*)(p + 12);
  v16h r;
  r[0]=(_Float16)a.x;  r[1]=(_Float16)a.y;  r[2]=(_Float16)a.z;  r[3]=(_Float16)a.w;
  r[4]=(_Float16)b.x;  r[5]=(_Float16)b.y;  r[6]=(_Float16)b.z;  r[7]=(_Float16)b.w;
  r[8]=(_Float16)c.x;  r[9]=(_Float16)c.y;  r[10]=(_Float16)c.z; r[11]=(_Float16)c.w;
  r[12]=(_Float16)d.x; r[13]=(_Float16)d.y; r[14]=(_Float16)d.z; r[15]=(_Float16)d.w;
  return r;
}

// A-fragment for one 16x32 chunk. rowk = &A[row][k0].
// lanes 0-15 hold K = 0..15, lanes 16-31 hold K = 16..31.
__device__ __forceinline__ v16h ldA32(const float* __restrict__ rowk, int hi) {
  const float* p = rowk + hi * 16;
  const float4 a = *(const float4*)(p + 0);
  const float4 b = *(const float4*)(p + 4);
  const float4 c = *(const float4*)(p + 8);
  const float4 d = *(const float4*)(p + 12);
  v16h r;
  r[0]=(_Float16)a.x;  r[1]=(_Float16)a.y;  r[2]=(_Float16)a.z;  r[3]=(_Float16)a.w;
  r[4]=(_Float16)b.x;  r[5]=(_Float16)b.y;  r[6]=(_Float16)b.z;  r[7]=(_Float16)b.w;
  r[8]=(_Float16)c.x;  r[9]=(_Float16)c.y;  r[10]=(_Float16)c.z; r[11]=(_Float16)c.w;
  r[12]=(_Float16)d.x; r[13]=(_Float16)d.y; r[14]=(_Float16)d.z; r[15]=(_Float16)d.w;
  return r;
}

// ---- generic C[M,N] = A[M,K] * W[N,K]^T + bias[N]  (one 16x16 tile per wave) --
__global__ __launch_bounds__(256) void wmma_gemm_bias(
    const float* __restrict__ A, const float* __restrict__ W,
    const float* __restrict__ bias, float* __restrict__ C,
    int M, int N, int K, int tiles_total) {
  const int lane = threadIdx.x & 31;
  const int gw   = blockIdx.x * 8 + (threadIdx.x >> 5);
  if (gw >= tiles_total) return;                 // wave-uniform: EXEC stays all-1s
  const int ntn  = N >> 4;
  const int mt   = gw / ntn, nt = gw % ntn;
  const int lrow = lane & 15, hi = lane >> 4;

  const float* arow = A + (size_t)(mt * 16 + lrow) * K;
  const float* wrow = W + (size_t)(nt * 16 + lrow) * K + hi * 16;

  v8f acc = {};
  for (int k0 = 0; k0 < K; k0 += 32) {
    __builtin_prefetch(arow + k0 + 64, 0, 0);    // global_prefetch_b8
    v16h a = ldA32(arow + k0, hi);
    v16h b = ld16f(wrow + k0);
    acc = WMMA_F16(a, b, acc);
  }
  const int n  = nt * 16 + lrow;
  const float bv = bias[n];
  float* crow = C + (size_t)(mt * 16 + 8 * hi) * N + n;
#pragma unroll
  for (int r = 0; r < 8; ++r) crow[(size_t)r * N] = acc[r] + bv;
}

// ---- edge scatter-add: agg[dst] += h[src]  (64 lanes/edge, float4 gathers) ----
__global__ __launch_bounds__(256) void scatter_add_kernel(
    const float* __restrict__ h, const int* __restrict__ ei,
    float* __restrict__ agg) {
  const unsigned tid = blockIdx.x * 256u + threadIdx.x;
  const unsigned e = tid >> 6;
  if (e >= NEDGE) return;
  const unsigned q = (tid & 63u) << 2;
  const int src = ei[e];
  const int dst = ei[NEDGE + e];
  const float4 v = *(const float4*)(h + (size_t)src * HID + q);
  float* d = agg + (size_t)dst * HID + q;
  unsafeAtomicAdd(d + 0, v.x);   // native global_atomic_add_f32, no CAS loop
  unsafeAtomicAdd(d + 1, v.y);
  unsafeAtomicAdd(d + 2, v.z);
  unsafeAtomicAdd(d + 3, v.w);
}

// ---- h = leaky_relu(agg * sigmoid(plast[i]) * sigmoid(syn[i])) ----------------
__global__ __launch_bounds__(256) void scale_leaky_kernel(
    const float* __restrict__ agg, float* __restrict__ h,
    const float* __restrict__ plast, const float* __restrict__ syn, int layer) {
  const int i = blockIdx.x * 256 + threadIdx.x;
  if (i >= N_NODES * HID) return;
  const float s = (1.f / (1.f + __expf(-plast[layer]))) *
                  (1.f / (1.f + __expf(-syn[layer])));
  const float v = agg[i] * s;
  h[i] = v > 0.f ? v : 0.01f * v;
}

// ---- flash attention ---------------------------------------------------------
// Block = 8 waves = 128 queries of ONE head. Each 32-key K/V tile is staged once
// per block into LDS with gfx1250 async global->LDS copies (ASYNCcnt-tracked),
// then consumed by all 8 waves -> 8x less L2 read traffic than per-wave streaming.
__global__ __launch_bounds__(256) void flash_attn_kernel(
    const float* __restrict__ qkv, float* __restrict__ obuf) {
  __shared__ float    ldsK[32 * 64];             // 8 KB: K tile [32 keys][64 dims]
  __shared__ float    ldsV[32 * 64];             // 8 KB: V tile [32 keys][64 dims]
  __shared__ _Float16 pbuf[8][512];              // 8 KB: per-wave 16x32 f16 P-tile

  const int lane = threadIdx.x & 31;
  const int wv   = threadIdx.x >> 5;
  const int head = blockIdx.x >> 6;              // 64 blocks of 128 queries per head
  const int qt   = (blockIdx.x & 63) * 8 + wv;   // this wave's 16-query tile
  const int lrow = lane & 15;
  const int hi   = lane >> 4;
  const int hoff = head * HEAD_DIM;

  // per-thread slice of the cooperative tile copy: 16B chunk c of row r16/r16+16
  const int c   = threadIdx.x & 15;              // 16-byte chunk within a 64-f row
  const int r16 = threadIdx.x >> 4;              // 0..15

  // Q fragments for the 16-row query tile (d=64 -> two 16x32 chunks), resident
  const float* qrow = qkv + (size_t)(qt * 16 + lrow) * QKV_DIM + hoff;
  const v16h aq0 = ldA32(qrow,      hi);
  const v16h aq1 = ldA32(qrow + 32, hi);

  float m[8], l[8];
  v8f o[4] = {{}, {}, {}, {}};
#pragma unroll
  for (int r = 0; r < 8; ++r) { m[r] = -1e30f; l[r] = 0.f; }

  for (int kb = 0; kb < N_NODES; kb += 32) {
    // ---- async global->LDS staging of K/V tiles (4 x b128 per thread) --------
#pragma unroll
    for (int half = 0; half < 2; ++half) {
      const int row = half * 16 + r16;           // key row within tile
      const size_t gro = (size_t)(kb + row) * QKV_DIM + hoff + c * 4;
      {
        const unsigned la = (unsigned)(size_t)&ldsK[row * 64 + c * 4];
        const unsigned long long ga = (unsigned long long)(size_t)(qkv + gro + HID);
        asm volatile("global_load_async_to_lds_b128 %0, %1, off"
                     :: "v"(la), "v"(ga) : "memory");
      }
      {
        const unsigned la = (unsigned)(size_t)&ldsV[row * 64 + c * 4];
        const unsigned long long ga = (unsigned long long)(size_t)(qkv + gro + 2 * HID);
        asm volatile("global_load_async_to_lds_b128 %0, %1, off"
                     :: "v"(la), "v"(ga) : "memory");
      }
    }
    asm volatile("s_wait_asynccnt 0x0" ::: "memory");  // own copies landed
    __syncthreads();                                   // all waves' copies landed

    // ---- S = Q * K_tile^T (two 16-key subtiles, K-dim 64 = 2 wmma each) ------
    const float* k0p = &ldsK[lrow * 64 + hi * 16];
    const float* k1p = k0p + 16 * 64;
    v8f s0 = {}, s1 = {};
    {
      v16h b00 = ld16f(k0p);        v16h b10 = ld16f(k1p);
      s0 = WMMA_F16(aq0, b00, s0);  s1 = WMMA_F16(aq0, b10, s1);
      v16h b01 = ld16f(k0p + 32);   v16h b11 = ld16f(k1p + 32);
      s0 = WMMA_F16(aq1, b01, s0);  s1 = WMMA_F16(aq1, b11, s1);
    }

    // ---- online softmax; D-layout: VGPR r <-> row r+8*hi, 16-lane half reduce -
#pragma unroll
    for (int r = 0; r < 8; ++r) {
      const float a0 = s0[r] * 0.125f;           // 1/sqrt(64)
      const float a1 = s1[r] * 0.125f;
      float tmax = fmaxf(a0, a1);
#pragma unroll
      for (int msk = 1; msk < 16; msk <<= 1)
        tmax = fmaxf(tmax, __shfl_xor(tmax, msk, 32));
      const float mn    = fmaxf(m[r], tmax);
      const float alpha = __expf(m[r] - mn);
      const float p0 = __expf(a0 - mn);
      const float p1 = __expf(a1 - mn);
      float rs = p0 + p1;
#pragma unroll
      for (int msk = 1; msk < 16; msk <<= 1)
        rs += __shfl_xor(rs, msk, 32);
      m[r] = mn;
      l[r] = l[r] * alpha + rs;
      o[0][r] *= alpha; o[1][r] *= alpha; o[2][r] *= alpha; o[3][r] *= alpha;
      pbuf[wv][(r + 8 * hi) * 32 + lrow]      = (_Float16)p0;
      pbuf[wv][(r + 8 * hi) * 32 + 16 + lrow] = (_Float16)p1;
    }

    // ---- reload P in A-fragment layout (same-wave DS ops are in-order) -------
    v16h pa;
    {
      const _Float16* pp = &pbuf[wv][lrow * 32 + hi * 16];
#pragma unroll
      for (int j = 0; j < 8; ++j) { pa[j] = pp[j]; pa[8 + j] = pp[8 + j]; }
    }

    // ---- O += P * V_tile (V: 32 keys x 64 dims -> 4 N-tiles of 16) -----------
#pragma unroll
    for (int t = 0; t < 4; ++t) {
      v16h bv;
      const float* vp = &ldsV[(hi * 16) * 64 + t * 16 + lrow];
#pragma unroll
      for (int j = 0; j < 16; ++j) bv[j] = (_Float16)vp[j * 64];
      o[t] = WMMA_F16(pa, bv, o[t]);
    }

    __syncthreads();                             // tile consumed; safe to overwrite
  }

  // ---- normalize and store O (concatenated heads layout [N, HID]) ------------
#pragma unroll
  for (int r = 0; r < 8; ++r) {
    const float inv = 1.f / l[r];
    float* dst = obuf + (size_t)(qt * 16 + r + 8 * hi) * HID + hoff + lrow;
    dst[0]  = o[0][r] * inv;
    dst[16] = o[1][r] * inv;
    dst[32] = o[2][r] * inv;
    dst[48] = o[3][r] * inv;
  }
}

extern "C" void kernel_launch(void* const* d_in, const int* in_sizes, int n_in,
                              void* d_out, int out_size, void* d_ws, size_t ws_size,
                              hipStream_t stream) {
  const float* x     = (const float*)d_in[0];
  const int*   ei    = (const int*)  d_in[1];   // JAX default x64-off -> int32
  const float* W_in  = (const float*)d_in[2];
  const float* b_in  = (const float*)d_in[3];
  const float* plast = (const float*)d_in[4];
  const float* syn   = (const float*)d_in[5];
  const float* inw   = (const float*)d_in[6];
  const float* inb   = (const float*)d_in[7];
  const float* outw  = (const float*)d_in[8];
  const float* outb  = (const float*)d_in[9];
  const float* Wout  = (const float*)d_in[10];
  const float* bout  = (const float*)d_in[11];
  float* out = (float*)d_out;

  float* h    = (float*)d_ws;                       // [N, HID]
  float* agg  = h    + (size_t)N_NODES * HID;       // [N, HID]
  float* qkvb = agg  + (size_t)N_NODES * HID;       // [N, 3*HID]
  float* ob   = qkvb + (size_t)N_NODES * QKV_DIM;   // [N, HID]
  float* pj   = ob   + (size_t)N_NODES * HID;       // [N, HID]

  // 1) h = x @ W_in^T + b_in       (512 x 16 = 8192 tiles)
  wmma_gemm_bias<<<8192 / 8, 256, 0, stream>>>(x, W_in, b_in, h,
                                               N_NODES, HID, IN_DIM, 8192);
  // 2) 3 graph layers: zero -> scatter-add -> scale+leaky_relu
  for (int i = 0; i < 3; ++i) {
    hipMemsetAsync(agg, 0, (size_t)N_NODES * HID * sizeof(float), stream);
    scatter_add_kernel<<<(NEDGE * 64) / 256, 256, 0, stream>>>(h, ei, agg);
    scale_leaky_kernel<<<(N_NODES * HID) / 256, 256, 0, stream>>>(agg, h, plast, syn, i);
  }
  // 3) qkv = h @ in_proj_w^T + in_proj_b   (512 x 48 = 24576 tiles)
  wmma_gemm_bias<<<24576 / 8, 256, 0, stream>>>(h, inw, inb, qkvb,
                                                N_NODES, QKV_DIM, HID, 24576);
  // 4) multi-head attention (flash, cooperative LDS tiles), o -> ob
  flash_attn_kernel<<<256, 256, 0, stream>>>(qkvb, ob);
  // 5) pj = ob @ out_proj_w^T + out_proj_b (8192 tiles)
  wmma_gemm_bias<<<8192 / 8, 256, 0, stream>>>(ob, outw, outb, pj,
                                               N_NODES, HID, HID, 8192);
  // 6) out = pj @ W_out^T + b_out          (512 x 8 = 4096 tiles)
  wmma_gemm_bias<<<4096 / 8, 256, 0, stream>>>(pj, Wout, bout, out,
                                               N_NODES, OUT_DIM, HID, 4096);
}
